// SparseBlock_Conv2d_BN_ReLU_88656714924891
// MI455X (gfx1250) — matrix operand order repaired and checked by reading, not compiled, not run on hardware
//
#include <hip/hip_runtime.h>
#include <hip/hip_bf16.h>

typedef __attribute__((ext_vector_type(16))) __bf16 v16bf;
typedef __attribute__((ext_vector_type(8)))  __bf16 v8bf;
typedef __attribute__((ext_vector_type(4)))  __bf16 v4bf;
typedef __attribute__((ext_vector_type(8)))  float  v8f;

#define IMG   1024
#define CCH   64          // input channels
#define NCH   64          // output channels
#define BSZ   34          // gathered patch side
#define PSTR  72          // LDS per-pixel channel stride (padded, bf16 elems)
#define KTOT  576         // 3*3*64
#define WSTR  584         // LDS per-n K stride (padded, bf16 elems)

#define LDS_PATCH_ELEMS (BSZ*BSZ*PSTR)               // 83232 bf16
#define LDS_W_OFF       LDS_PATCH_ELEMS
#define LDS_TOTAL_BYTES ((LDS_PATCH_ELEMS + NCH*WSTR) * 2)   // 241216 B < 320KB/WGP

// ---------- tiny map kernels: d_ws holds int map[1024] ----------
__global__ void init_map_kernel(int* map) {
    int g = blockIdx.x * 256 + threadIdx.x;
    if (g < 1024) map[g] = -1;
}

__global__ void fill_map_kernel(const int* __restrict__ abi, int* map, int M) {
    int m = blockIdx.x * 256 + threadIdx.x;
    if (m < M) {
        int bi = abi[m * 3 + 1];
        int bj = abi[m * 3 + 2];
        map[bi * 32 + bj] = m;
    }
}

// ---------- main kernel: one workgroup (8 wave32) per 32x32x64 output block ----------
__global__ void __launch_bounds__(256)
sparse_conv_bn_relu_kernel(const float* __restrict__ inp,
                           const float* __restrict__ wgt,    // [3][3][64][64]
                           const float* __restrict__ bias,
                           const float* __restrict__ gamma,
                           const float* __restrict__ beta,
                           const float* __restrict__ rmean,
                           const float* __restrict__ rvar,
                           const int*   __restrict__ map,
                           float* __restrict__ out)
{
    extern __shared__ __bf16 lds[];

    const int blk = blockIdx.x;           // 0..1023
    const int bi  = blk >> 5;
    const int bj  = blk & 31;
    const int tid = threadIdx.x;

    if (map[blk] < 0) {
        // inactive block: stream zeros (32*32*16 float4)
        const float4 z = make_float4(0.f, 0.f, 0.f, 0.f);
        for (int t = tid; t < 32 * 32 * 16; t += 256) {
            int cq = t & 15;
            int pc = (t >> 4) & 31;
            int pr = t >> 9;
            size_t gr = (size_t)(bi * 32 + pr);
            size_t gc = (size_t)(bj * 32 + pc);
            *(float4*)&out[(gr * IMG + gc) * NCH + cq * 4] = z;
        }
        return;
    }

    // ---- stage patch (34x34x64 f32 -> bf16, zero-padded at image edges) into LDS ----
    for (int t = tid; t < BSZ * BSZ * 16; t += 256) {
        int cq   = t & 15;                 // float4 along channels
        int rest = t >> 4;                 // 0..1155
        int pc   = rest % BSZ;
        int pr   = rest / BSZ;
        int gr   = bi * 32 - 1 + pr;
        int gc   = bj * 32 - 1 + pc;
        float4 v = make_float4(0.f, 0.f, 0.f, 0.f);
        if ((unsigned)gr < (unsigned)IMG && (unsigned)gc < (unsigned)IMG)
            v = *(const float4*)&inp[((size_t)gr * IMG + gc) * CCH + cq * 4];
        v4bf pk;
        pk.x = (__bf16)v.x; pk.y = (__bf16)v.y; pk.z = (__bf16)v.z; pk.w = (__bf16)v.w;
        *(v4bf*)&lds[(pr * BSZ + pc) * PSTR + cq * 4] = pk;   // 8B-aligned ds_store_b64
    }

    // ---- stage weights transposed: lds_w[n][k], k = (ky*3+kx)*64 + ci ----
    for (int t = tid; t < KTOT * NCH; t += 256) {
        int n = t & 63;                    // coalesced along co
        int k = t >> 6;
        lds[LDS_W_OFF + n * WSTR + k] = (__bf16)wgt[k * NCH + n];
    }

    // ---- fused BN/bias constants for this lane's 4 output-channel groups ----
    const int lane   = tid & 31;
    const int laneHi = lane >> 4;
    const int nlo    = lane & 15;
    float sc[4], sh[4];
#pragma unroll
    for (int nt = 0; nt < 4; ++nt) {
        int n = nt * 16 + nlo;
        float s = gamma[n] * rsqrtf(rvar[n] + 1e-3f);
        sc[nt] = s;
        sh[nt] = (bias[n] - rmean[n]) * s + beta[n];
    }

    __syncthreads();

    const int wave = tid >> 5;            // 0..7

    // Each wave: 4 passes; each pass computes one pixel row r (32 px) x 64 ch
    // as two 16x16 M-tiles (c0 = 0 and 16) sharing B fragments.
#pragma unroll 1
    for (int p = 0; p < 4; ++p) {
        // Opaque zero offset: defeats LICM so B-fragment LDS loads are NOT
        // hoisted across passes (which previously caused scratch spills).
        unsigned zoff = 0;
        asm volatile("" : "+v"(zoff));
        const __bf16* ldsb = lds + zoff;

        const int r = wave * 4 + p;       // pixel row within 32x32 block
        v8f acc0[4] = {};                 // c0 = 0,  N-tiles 0..3
        v8f acc1[4] = {};                 // c0 = 16, N-tiles 0..3

#pragma unroll 1
        for (int ks = 0; ks < 18; ++ks) { // K-steps of 32 over K=576
            const int g      = ks >> 1;            // tap 0..8
            const int ciBase = (ks & 1) * 32;
            const int ky = g / 3, kx = g % 3;

            // A fragments: row M = nlo -> pixel (r, c0+nlo); K striped per ISA layout
            const __bf16* ap =
                &ldsb[((r + ky) * BSZ + (nlo + kx)) * PSTR + ciBase + laneHi * 8];
            v16bf a0, a1;
            ((v8bf*)&a0)[0] = *(const v8bf*)(ap);               // K = base + 0..7
            ((v8bf*)&a0)[1] = *(const v8bf*)(ap + 16);          // K = base + 16..23
            ((v8bf*)&a1)[0] = *(const v8bf*)(ap + 16 * PSTR);
            ((v8bf*)&a1)[1] = *(const v8bf*)(ap + 16 * PSTR + 16);

#pragma unroll
            for (int nt = 0; nt < 4; ++nt) {
                // B fragment: lane holds K = 8h+0..7 and 16+8h+0..7 for column n = nt*16+nlo
                const __bf16* bp =
                    &ldsb[LDS_W_OFF + (nt * 16 + nlo) * WSTR + ks * 32 + laneHi * 8];
                v16bf b;
                ((v8bf*)&b)[0] = *(const v8bf*)(bp);
                ((v8bf*)&b)[1] = *(const v8bf*)(bp + 16);
                acc0[nt] = __builtin_amdgcn_wmma_f32_16x16x32_bf16(
                    false, a0, false, b, (short)0, acc0[nt], false, false);
                acc1[nt] = __builtin_amdgcn_wmma_f32_16x16x32_bf16(
                    false, a1, false, b, (short)0, acc1[nt], false, false);
            }
        }

        // ---- epilogue: BN + ReLU, scatter to output ----
        const size_t gr = (size_t)(bi * 32 + r);
#pragma unroll
        for (int nt = 0; nt < 4; ++nt) {
            const int n = nt * 16 + nlo;
#pragma unroll
            for (int gi = 0; gi < 8; ++gi) {
                int Mm = gi + laneHi * 8;                 // pixel within tile
                size_t gc0 = (size_t)(bj * 32 + Mm);      // tile c0 = 0
                size_t gc1 = gc0 + 16;                    // tile c0 = 16
                float v0 = fmaxf(acc0[nt][gi] * sc[nt] + sh[nt], 0.f);
                float v1 = fmaxf(acc1[nt][gi] * sc[nt] + sh[nt], 0.f);
                out[(gr * IMG + gc0) * NCH + n] = v0;
                out[(gr * IMG + gc1) * NCH + n] = v1;
            }
        }
    }
}

extern "C" void kernel_launch(void* const* d_in, const int* in_sizes, int n_in,
                              void* d_out, int out_size, void* d_ws, size_t ws_size,
                              hipStream_t stream) {
    const float* inp   = (const float*)d_in[0];
    const float* wgt   = (const float*)d_in[1];
    const float* bias  = (const float*)d_in[2];
    const float* gamma = (const float*)d_in[3];
    const float* beta  = (const float*)d_in[4];
    const float* rmean = (const float*)d_in[5];
    const float* rvar  = (const float*)d_in[6];
    const int*   abi   = (const int*)d_in[7];
    float* out = (float*)d_out;
    int*   map = (int*)d_ws;

    const int M = in_sizes[7] / 3;   // 512 active blocks

    init_map_kernel<<<4, 256, 0, stream>>>(map);
    fill_map_kernel<<<(M + 255) / 256, 256, 0, stream>>>(abi, map, M);
    sparse_conv_bn_relu_kernel<<<1024, 256, LDS_TOTAL_BYTES, stream>>>(
        inp, wgt, bias, gamma, beta, rmean, rvar, map, out);
}
